// CNN_75960791597285
// MI455X (gfx1250) — matrix operand order repaired and checked by reading, not compiled, not run on hardware
//
#include <hip/hip_runtime.h>
#include <hip/hip_bf16.h>
#include <cstdint>
#include <cstddef>

// ---------------------------------------------------------------------------
// Types for CDNA5 WMMA / TDM
// ---------------------------------------------------------------------------
typedef __attribute__((ext_vector_type(16))) __bf16       v16bf;
typedef __attribute__((ext_vector_type(8)))  float        v8f;
typedef __attribute__((ext_vector_type(4)))  unsigned int v4u;
typedef __attribute__((ext_vector_type(8)))  int          v8i_t;
typedef __attribute__((ext_vector_type(4)))  int          v4i_t;

#define BN_EPS 1e-5f

// B tile in LDS: 32 rows x 64 bf16, TDM-padded 1 dword per 32 dwords
// -> row stride 33 dwords = 66 bf16 (bank-conflict-free fragment reads).
#define BS_ROW   66
#define BS_ELEMS (32 * BS_ROW)

// ---------------------------------------------------------------------------
// TDM: DMA one 32(K) x 64(N) bf16 tile of the [K][Cout] weight matrix into
// LDS at ldsOff, with pad_enable so each 128B row gets +4B of LDS padding.
// D# layout per CDNA5 ISA.
// ---------------------------------------------------------------------------
__device__ __forceinline__ void tdm_load_b_tile(const __bf16* gsrc,
                                                unsigned ldsOff, int Cout)
{
    unsigned long long ga = (unsigned long long)(size_t)gsrc;
    v4u g0;
    g0[0] = 1u;                                   // count=1 (valid), no gather
    g0[1] = ldsOff;                               // lds_addr (bytes)
    g0[2] = (unsigned)(ga & 0xffffffffu);         // global_addr[31:0]
    g0[3] = (unsigned)((ga >> 32) & 0x01ffffffu)  // global_addr[56:32]
          | (2u << 30);                           // type = 2 ("image")
    v8i_t g1;
    g1[0] = (1 << 16)                             // data_size = 1 (2 bytes)
          | (1 << 20)                             // pad_enable
          | (4 << 22);                            // pad_interval: 32 dwords
                                                  // pad_amount = 0 -> 1 dword
    g1[1] = (int)(64u << 16);                     // tensor_dim0[15:0] = 64
    g1[2] = (int)(32u << 16);                     // tensor_dim1[15:0] = 32
    g1[3] = (int)(64u << 16);                     // tile_dim0 = 64
    g1[4] = 32;                                   // tile_dim1 = 32, tile_dim2=0
    g1[5] = Cout;                                 // tensor_dim0_stride[31:0]
    g1[6] = 0;
    g1[7] = 0;
    v4i_t z4 = {0, 0, 0, 0};
#if defined(__clang_major__) && (__clang_major__ >= 23)
    v8i_t z8 = {0, 0, 0, 0, 0, 0, 0, 0};
    __builtin_amdgcn_tensor_load_to_lds(g0, g1, z4, z4, z8, 0);
#else
    __builtin_amdgcn_tensor_load_to_lds(g0, g1, z4, z4, 0);
#endif
}

// ---------------------------------------------------------------------------
// BN fold: scale = g*rsqrt(v+eps); bias = beta + (conv_bias - mean)*scale
// ---------------------------------------------------------------------------
__global__ void bn_fold_kernel(const float* __restrict__ g,  const float* __restrict__ be,
                               const float* __restrict__ m,  const float* __restrict__ v,
                               const float* __restrict__ cb, float* __restrict__ scale,
                               float* __restrict__ bias, int C)
{
    int i = blockIdx.x * blockDim.x + threadIdx.x;
    if (i >= C) return;
    float s = g[i] * rsqrtf(v[i] + BN_EPS);
    scale[i] = s;
    float c = cb ? cb[i] : 0.0f;
    bias[i] = be[i] + (c - m[i]) * s;
}

// ---------------------------------------------------------------------------
// Weight transpose+convert: src fp32 [N, K] -> dst bf16 [K, Npad], zero pad.
// ---------------------------------------------------------------------------
__global__ void wtrans_kernel(const float* __restrict__ w, unsigned short* __restrict__ dst,
                              int K, int N, int Npad)
{
    long idx = (long)blockIdx.x * blockDim.x + threadIdx.x;
    long tot = (long)K * Npad;
    if (idx >= tot) return;
    int k = (int)(idx / Npad);
    int n = (int)(idx % Npad);
    float val = (n < N) ? w[(size_t)n * K + k] : 0.0f;
    ((__bf16*)dst)[idx] = (__bf16)val;
}

// ---------------------------------------------------------------------------
// conv1: 7x7 stride 2 pad 3, Cin=3 (K=147, not a multiple of 32 -> scalar),
// fused BN + ReLU, output bf16 NCHW (B,64,112,112)
// ---------------------------------------------------------------------------
__global__ void conv1_bn_relu_kernel(const float* __restrict__ x, const float* __restrict__ w,
                                     const float* __restrict__ scale, const float* __restrict__ bias,
                                     unsigned short* __restrict__ out_, int B)
{
    size_t idx = (size_t)blockIdx.x * blockDim.x + threadIdx.x;
    size_t total = (size_t)B * 64 * 112 * 112;
    if (idx >= total) return;
    int ow = (int)(idx % 112); size_t r = idx / 112;
    int oh = (int)(r % 112);   r /= 112;
    int co = (int)(r % 64);
    int b  = (int)(r / 64);
    float acc = 0.0f;
    for (int ci = 0; ci < 3; ci++) {
        const float* xp = x + (((size_t)b * 3 + ci) * 224) * 224;
        const float* wp = w + (((size_t)co * 3 + ci) * 7) * 7;
        for (int kh = 0; kh < 7; kh++) {
            int ih = oh * 2 + kh - 3;
            if (ih < 0 || ih >= 224) continue;
            for (int kw = 0; kw < 7; kw++) {
                int iw = ow * 2 + kw - 3;
                if (iw < 0 || iw >= 224) continue;
                acc += xp[ih * 224 + iw] * wp[kh * 7 + kw];
            }
        }
    }
    float v = fmaxf(acc * scale[co] + bias[co], 0.0f);
    ((__bf16*)out_)[idx] = (__bf16)v;
}

// ---------------------------------------------------------------------------
// maxpool 3x3 stride 3 VALID: (B,64,112,112) -> (B,64,37,37), bf16
// ---------------------------------------------------------------------------
__global__ void maxpool3_kernel(const unsigned short* __restrict__ in_,
                                unsigned short* __restrict__ out_, int B)
{
    size_t idx = (size_t)blockIdx.x * blockDim.x + threadIdx.x;
    size_t total = (size_t)B * 64 * 37 * 37;
    if (idx >= total) return;
    const __bf16* in = (const __bf16*)in_;
    int ow = (int)(idx % 37);
    int oh = (int)((idx / 37) % 37);
    size_t bc = idx / (37 * 37);
    const __bf16* p = in + bc * (size_t)(112 * 112);
    float m = -3.0e38f;
    for (int kh = 0; kh < 3; kh++)
        for (int kw = 0; kw < 3; kw++)
            m = fmaxf(m, (float)p[(oh * 3 + kh) * 112 + (ow * 3 + kw)]);
    ((__bf16*)out_)[idx] = (__bf16)m;
}

// ---------------------------------------------------------------------------
// SPP over (B,512,5,5): levels 5 (identity), 2 (bins [0,3),[2,5)), 1 (global).
// Feature layout: feat[b, c*30 + j], bf16.
// ---------------------------------------------------------------------------
__global__ void spp_kernel(const unsigned short* __restrict__ in_,
                           unsigned short* __restrict__ out_, int B)
{
    int idx = blockIdx.x * blockDim.x + threadIdx.x;
    if (idx >= B * 512) return;
    const __bf16* in = (const __bf16*)in_ + (size_t)idx * 25;
    float v[25];
    float tot = 0.0f;
#pragma unroll
    for (int i = 0; i < 25; i++) { v[i] = (float)in[i]; tot += v[i]; }
    int b = idx >> 9;
    int c = idx & 511;
    __bf16* o = (__bf16*)out_ + (size_t)b * 15360 + (size_t)c * 30;
#pragma unroll
    for (int i = 0; i < 25; i++) o[i] = (__bf16)v[i];
    const int lo[2] = {0, 2}, hi[2] = {3, 5};
#pragma unroll
    for (int bi = 0; bi < 2; bi++)
#pragma unroll
        for (int bj = 0; bj < 2; bj++) {
            float s = 0.0f; int cnt = 0;
            for (int rr = lo[bi]; rr < hi[bi]; rr++)
                for (int cc = lo[bj]; cc < hi[bj]; cc++) { s += v[rr * 5 + cc]; cnt++; }
            o[25 + bi * 2 + bj] = (__bf16)(s / (float)cnt);
        }
    o[29] = (__bf16)(tot / 25.0f);
}

// ---------------------------------------------------------------------------
// Implicit-GEMM conv / GEMM with v_wmma_f32_16x16x32_bf16.
//   A: im2col of bf16 activations (NCHW)  [M = B*Ho*Wo, K = Cin*KH*KW]
//   B: bf16 weights [K][Cout], staged to LDS by the Tensor Data Mover with
//      double buffering (TENSORcnt) and TDM LDS padding for conflict-free
//      fragment reads.
//   D: per-channel scale/bias (BN fold), optional residual add, optional ReLU,
//      output bf16 NCHW or fp32 (fc3).
// Block tile 128(M) x 64(N), K step 32, 8 waves, wave tile 32x32 = 2x2 frags.
// Requires: K % 32 == 0, Cout % 64 == 0 (guaranteed by the caller).
// ---------------------------------------------------------------------------
__global__ __launch_bounds__(256)
void conv_wmma_kernel(const unsigned short* __restrict__ act_,
                      const unsigned short* __restrict__ wB_,
                      const float* __restrict__ scale,
                      const float* __restrict__ bias,
                      const unsigned short* __restrict__ res_,
                      void* __restrict__ out_,
                      int M, int K, int Cout, int CoutStore,
                      int Cin, int Hin, int Win, int Ho, int Wo,
                      int KH, int KW, int stride, int pad,
                      int relu, int outF32)
{
    const __bf16* act = (const __bf16*)act_;
    const __bf16* wB  = (const __bf16*)wB_;
    const __bf16* res = (const __bf16*)res_;

    __shared__ __bf16 As[128 * 40];        // 128 M-rows x 32 K (+pad)
    __shared__ __bf16 Bs[2][BS_ELEMS];     // double-buffered TDM destination

    const int t    = threadIdx.x;
    const int lane = t & 31;
    const int wv   = t >> 5;               // wave 0..7
    const int wm   = wv & 3;               // M direction
    const int wn   = wv >> 2;              // N direction
    const int hh   = lane >> 4;            // lane half
    const int l15  = lane & 15;

    const int mBlock = blockIdx.x * 128;
    const int nBlock = blockIdx.y * 64;

    const int HW  = Ho * Wo;
    const int KHW = KH * KW;

    // cooperative A-load mapping: one (row, 16-wide k group) per thread
    const int arow = t & 127;
    const int kgrp = t >> 7;
    int am = mBlock + arow; if (am >= M) am = M - 1;
    const int ab  = am / HW;
    const int ar  = am - ab * HW;
    const int aoh = ar / Wo;
    const int aow = ar - aoh * Wo;
    const int ihBase = aoh * stride - pad;
    const int iwBase = aow * stride - pad;

    v8f acc[2][2];
#pragma unroll
    for (int i = 0; i < 2; i++)
#pragma unroll
        for (int j = 0; j < 2; j++)
#pragma unroll
            for (int e = 0; e < 8; e++) acc[i][j][e] = 0.0f;

    const unsigned bsOff[2] = { (unsigned)(size_t)(void*)&Bs[0][0],
                                (unsigned)(size_t)(void*)&Bs[1][0] };

    // prologue: wave 0 kicks off the first B-tile DMA
    if (wv == 0)
        tdm_load_b_tile(wB + (size_t)0 * Cout + nBlock, bsOff[0], Cout);

    const int nTiles = K >> 5;
    for (int ti = 0; ti < nTiles; ti++) {
        const int kb  = ti << 5;
        const int buf = ti & 1;

        __syncthreads();   // previous iteration done reading As
        // ---- stage A tile (im2col gather, zero pad) ----
#pragma unroll 4
        for (int j = 0; j < 16; j++) {
            int k  = kb + kgrp * 16 + j;
            int ci = k / KHW;
            int rr = k - ci * KHW;
            int kh = rr / KW;
            int kw = rr - kh * KW;
            int ih = ihBase + kh;
            int iw = iwBase + kw;
            __bf16 v = (__bf16)0.0f;
            if (ih >= 0 && ih < Hin && iw >= 0 && iw < Win)
                v = act[(((size_t)ab * Cin + ci) * Hin + ih) * Win + iw];
            As[arow * 40 + kgrp * 16 + j] = v;
        }
        // wave 0: B-tile DMA for this step complete?
        if (wv == 0)
            __builtin_amdgcn_s_wait_tensorcnt(0);
        __syncthreads();   // publish As stores and TDM-written Bs[buf]

        // wave 0: launch DMA of the *next* B tile into the other buffer;
        // it overlaps with the WMMA work below.
        if (wv == 0 && (ti + 1) < nTiles)
            tdm_load_b_tile(wB + (size_t)(kb + 32) * Cout + nBlock,
                            bsOff[(ti + 1) & 1], Cout);

        // ---- fragments per ISA 7.12.2 layouts ----
        v16bf afrag[2], bfrag[2];
#pragma unroll
        for (int mf = 0; mf < 2; mf++) {
            int row = (wm * 32 + mf * 16 + l15) * 40;
#pragma unroll
            for (int d = 0; d < 8; d++) {
                int k0 = (d < 4 ? 2 * d : 16 + 2 * (d - 4)) + hh * 8;
                afrag[mf][2 * d]     = As[row + k0];
                afrag[mf][2 * d + 1] = As[row + k0 + 1];
            }
        }
#pragma unroll
        for (int nf = 0; nf < 2; nf++) {
            int base = wn * 32 + nf * 16 + l15;  // B: lane%16 == N column
#pragma unroll
            for (int p = 0; p < 8; p++) {
                int k0 = (p < 4 ? 2 * p : 16 + 2 * (p - 4)) + hh * 8;
                bfrag[nf][2 * p]     = Bs[buf][k0 * BS_ROW + base];
                bfrag[nf][2 * p + 1] = Bs[buf][(k0 + 1) * BS_ROW + base];
            }
        }
#pragma unroll
        for (int mf = 0; mf < 2; mf++)
#pragma unroll
            for (int nf = 0; nf < 2; nf++)
                acc[mf][nf] = __builtin_amdgcn_wmma_f32_16x16x32_bf16(
                    false, afrag[mf], false, bfrag[nf],
                    (short)0, acc[mf][nf], false, false);
    }

    // ---- epilogue: scale/bias (+residual) (+relu), store bf16 or f32 ----
#pragma unroll
    for (int nf = 0; nf < 2; nf++) {
        int nc = nBlock + wn * 32 + nf * 16 + l15;
        float s = 1.0f, bi = 0.0f;
        if (nc < Cout) {
            if (scale) s = scale[nc];
            if (bias)  bi = bias[nc];
        }
#pragma unroll
        for (int mf = 0; mf < 2; mf++) {
#pragma unroll
            for (int r = 0; r < 8; r++) {
                int m = mBlock + wm * 32 + mf * 16 + hh * 8 + r;
                if (m < M && nc < CoutStore) {
                    int b   = m / HW;
                    int pos = m - b * HW;
                    size_t oi = ((size_t)b * CoutStore + nc) * HW + pos;
                    float v = acc[mf][nf][r] * s + bi;
                    if (res_) v += (float)res[oi];
                    if (relu) v = fmaxf(v, 0.0f);
                    if (outF32) ((float*)out_)[oi] = v;
                    else        ((__bf16*)out_)[oi] = (__bf16)v;
                }
            }
        }
    }
}

// ---------------------------------------------------------------------------
// Host orchestration
// ---------------------------------------------------------------------------
extern "C" void kernel_launch(void* const* d_in, const int* in_sizes, int n_in,
                              void* d_out, int out_size, void* d_ws, size_t ws_size,
                              hipStream_t stream)
{
    (void)in_sizes; (void)n_in; (void)out_size; (void)ws_size;
    int ii = 0;
    auto nextp = [&]() -> const float* { return (const float*)d_in[ii++]; };

    const float* X   = nextp();
    const float* c1w = nextp();
    const float* c1b = nextp();
    const float* c1g = nextp(); const float* c1be = nextp();
    const float* c1m = nextp(); const float* c1v  = nextp();

    struct Blk {
        const float *w1,*b1,*g1,*be1,*m1,*v1;
        const float *w2,*b2,*g2,*be2,*m2,*v2;
        const float *scw,*scg,*scbe,*scm,*scv;
        int ci, co, st; bool sc;
    } blk[16];
    static const int cfg[16][3] = {
        {64,128,2},{128,128,1},{128,128,1},
        {128,256,2},{256,256,1},{256,256,1},{256,256,1},
        {256,512,2},{512,512,1},{512,512,1},{512,512,1},{512,512,1},{512,512,1},
        {512,512,1},{512,512,1},{512,512,1}};
    for (int i = 0; i < 16; i++) {
        Blk& b = blk[i];
        b.ci = cfg[i][0]; b.co = cfg[i][1]; b.st = cfg[i][2];
        b.sc = (b.st != 1) || (b.ci != b.co);
        b.w1 = nextp(); b.b1 = nextp();
        b.g1 = nextp(); b.be1 = nextp(); b.m1 = nextp(); b.v1 = nextp();
        b.w2 = nextp(); b.b2 = nextp();
        b.g2 = nextp(); b.be2 = nextp(); b.m2 = nextp(); b.v2 = nextp();
        if (b.sc) {
            b.scw = nextp();
            b.scg = nextp(); b.scbe = nextp(); b.scm = nextp(); b.scv = nextp();
        } else {
            b.scw = b.scg = b.scbe = b.scm = b.scv = nullptr;
        }
    }
    const float* fc1w = nextp(); const float* fc1b = nextp();
    const float* fc2w = nextp(); const float* fc2b = nextp();
    const float* fc3w = nextp(); const float* fc3b = nextp();

    // bump allocator over workspace
    char* wp = (char*)d_ws;
    auto alloc = [&](size_t bytes) -> void* {
        void* p = (void*)wp; wp += (bytes + 255) & ~(size_t)255; return p;
    };
    auto allocBf = [&](size_t n) { return (unsigned short*)alloc(n * 2); };
    auto allocF  = [&](size_t n) { return (float*)alloc(n * 4); };

    auto trans = [&](const float* w, unsigned short* dst, int K, int N, int Npad) {
        long tot = (long)K * Npad;
        wtrans_kernel<<<dim3((unsigned)((tot + 255) / 256)), dim3(256), 0, stream>>>(w, dst, K, N, Npad);
    };
    auto fold = [&](const float* g, const float* be, const float* m, const float* v,
                    const float* cb, float* sc, float* bi, int C) {
        bn_fold_kernel<<<dim3((C + 255) / 256), dim3(256), 0, stream>>>(g, be, m, v, cb, sc, bi, C);
    };
    auto conv = [&](const unsigned short* act, const unsigned short* wB,
                    const float* sc, const float* bi, const unsigned short* res,
                    void* out, int B, int Cin, int Hin, int Win,
                    int Cout, int CoutStore, int KH, int KW, int stride, int pad,
                    int relu, int outF32) {
        int Ho = (Hin + 2 * pad - KH) / stride + 1;
        int Wo = (Win + 2 * pad - KW) / stride + 1;
        int M = B * Ho * Wo;
        int K = Cin * KH * KW;
        dim3 grid((M + 127) / 128, Cout / 64);
        conv_wmma_kernel<<<grid, dim3(256), 0, stream>>>(
            act, wB, sc, bi, res, out, M, K, Cout, CoutStore,
            Cin, Hin, Win, Ho, Wo, KH, KW, stride, pad, relu, outF32);
    };

    // ---- weight conversion + BN folding (re-done every call: deterministic) ----
    float* s_c1 = allocF(64); float* bi_c1 = allocF(64);
    fold(c1g, c1be, c1m, c1v, c1b, s_c1, bi_c1, 64);

    unsigned short *wb1[16], *wb2[16], *wbsc[16];
    float *sc1[16], *bi1[16], *sc2[16], *bi2[16], *scS[16], *biS[16];
    for (int i = 0; i < 16; i++) {
        Blk& b = blk[i];
        int K1 = b.ci * 9, K2 = b.co * 9;
        wb1[i] = allocBf((size_t)K1 * b.co);
        wb2[i] = allocBf((size_t)K2 * b.co);
        trans(b.w1, wb1[i], K1, b.co, b.co);
        trans(b.w2, wb2[i], K2, b.co, b.co);
        sc1[i] = allocF(b.co); bi1[i] = allocF(b.co);
        sc2[i] = allocF(b.co); bi2[i] = allocF(b.co);
        fold(b.g1, b.be1, b.m1, b.v1, b.b1, sc1[i], bi1[i], b.co);
        fold(b.g2, b.be2, b.m2, b.v2, b.b2, sc2[i], bi2[i], b.co);
        if (b.sc) {
            wbsc[i] = allocBf((size_t)b.ci * b.co);
            trans(b.scw, wbsc[i], b.ci, b.co, b.co);
            scS[i] = allocF(b.co); biS[i] = allocF(b.co);
            fold(b.scg, b.scbe, b.scm, b.scv, nullptr, scS[i], biS[i], b.co);
        } else {
            wbsc[i] = nullptr; scS[i] = biS[i] = nullptr;
        }
    }
    unsigned short* wfc1 = allocBf((size_t)15360 * 4096);
    unsigned short* wfc2 = allocBf((size_t)4096 * 4096);
    unsigned short* wfc3 = allocBf((size_t)4096 * 1024);
    trans(fc1w, wfc1, 15360, 4096, 4096);
    trans(fc2w, wfc2, 4096, 4096, 4096);
    trans(fc3w, wfc3, 4096, 1000, 1024);

    // ---- activation buffers ----
    const int B = 32;
    unsigned short* aConv1 = allocBf((size_t)B * 64 * 112 * 112);
    unsigned short* aPool  = allocBf((size_t)B * 64 * 37 * 37);
    size_t ringElems = (size_t)B * 128 * 19 * 19;   // largest stage tensor
    unsigned short* ring[4];
    for (int i = 0; i < 4; i++) ring[i] = allocBf(ringElems);
    unsigned short* feat = allocBf((size_t)B * 15360);
    unsigned short* f1   = allocBf((size_t)B * 4096);
    unsigned short* f2   = allocBf((size_t)B * 4096);

    // ---- forward pass ----
    {
        size_t tot = (size_t)B * 64 * 112 * 112;
        conv1_bn_relu_kernel<<<dim3((unsigned)((tot + 255) / 256)), dim3(256), 0, stream>>>(
            X, c1w, s_c1, bi_c1, aConv1, B);
    }
    {
        size_t tot = (size_t)B * 64 * 37 * 37;
        maxpool3_kernel<<<dim3((unsigned)((tot + 255) / 256)), dim3(256), 0, stream>>>(
            aConv1, aPool, B);
    }

    unsigned short* cur = aPool;
    int C = 64, H = 37, W = 37;
    int rc = 0;
    for (int i = 0; i < 16; i++) {
        Blk& b = blk[i];
        int Ho = (H + 2 - 3) / b.st + 1;
        int Wo = (W + 2 - 3) / b.st + 1;
        unsigned short* t1  = ring[rc % 4];
        unsigned short* tsc = ring[(rc + 1) % 4];
        unsigned short* o   = ring[(rc + 2) % 4];
        rc += 3;
        // relu(bn1(conv1(x)))
        conv(cur, wb1[i], sc1[i], bi1[i], nullptr, t1, B, C, H, W, b.co, b.co, 3, 3, b.st, 1, 1, 0);
        const unsigned short* resid;
        if (b.sc) {  // bn_sc(conv_sc(x)) 1x1, no relu
            conv(cur, wbsc[i], scS[i], biS[i], nullptr, tsc, B, C, H, W, b.co, b.co, 1, 1, b.st, 0, 0, 0);
            resid = tsc;
        } else {
            resid = cur;
        }
        // relu(bn2(conv2(t1)) + resid)
        conv(t1, wb2[i], sc2[i], bi2[i], resid, o, B, b.co, Ho, Wo, b.co, b.co, 3, 3, 1, 1, 1, 0);
        cur = o; C = b.co; H = Ho; W = Wo;
    }

    // SPP over (B,512,5,5) -> (B,15360)
    spp_kernel<<<dim3((B * 512 + 255) / 256), dim3(256), 0, stream>>>(cur, feat, B);

    // FC head (GEMMs via the same WMMA kernel, H=W=1)
    conv(feat, wfc1, nullptr, fc1b, nullptr, f1, B, 15360, 1, 1, 4096, 4096, 1, 1, 1, 0, 1, 0);
    conv(f1,   wfc2, nullptr, fc2b, nullptr, f2, B, 4096,  1, 1, 4096, 4096, 1, 1, 1, 0, 1, 0);
    // fc3: N=1000 padded to 1024, fp32 output straight into d_out
    conv(f2,   wfc3, nullptr, fc3b, nullptr, d_out, B, 4096, 1, 1, 1024, 1000, 1, 1, 1, 0, 0, 1);
}
